// CEMSA_850403525284
// MI455X (gfx1250) — compile-verified, matrix operand order checked by reading, not run on hardware
//
#include <hip/hip_runtime.h>
#include <stdint.h>

#define Bn 4
#define Hh 64
#define Ww 64
#define Cc 256
#define HEADS 8
#define Dd 32
#define Nn (Hh*Ww)           // 4096
#define Mm ((Hh/2)*(Ww/2))   // 1024
#define SCALE 0.17677669529663687f
#define EPSf 1e-6f

typedef __bf16 bf16;
typedef __attribute__((ext_vector_type(16))) __bf16 v16bf;
typedef __attribute__((ext_vector_type(8)))  float  v8f;

// ---------------------------------------------------------------------------
// WMMA wrapper: D(16x16,f32) = A(16x32,bf16) * B(32x16,bf16) + C
__device__ __forceinline__ v8f wmma_bf16(v16bf a, v16bf b, v8f c) {
  return __builtin_amdgcn_wmma_f32_16x16x32_bf16(false, a, false, b, (short)0, c,
                                                 false, false);
}

// Load a 16-row x 32-col bf16 operand tile in the CDNA5 A/B register layout.
// lane 0-15  : row r0+lane,    K = k0+{0..7} and k0+{16..23}
// lane 16-31 : row r0+lane-16, K = k0+{8..15} and k0+{24..31}
// Each lane: two 16-byte contiguous loads (global_load_b128).
__device__ __forceinline__ v16bf load_tile(const bf16* __restrict__ base,
                                           int stride, int r0, int k0) {
  const int lane = threadIdx.x & 31;
  const bf16* p = base + (size_t)(r0 + (lane & 15)) * stride + k0 +
                  ((lane & 16) ? 8 : 0);
  union { v16bf v; uint4 q[2]; } t;
  t.q[0] = *reinterpret_cast<const uint4*>(p);
  t.q[1] = *reinterpret_cast<const uint4*>(p + 16);
  return t.v;
}

// Pack 8 f32 accumulator values (scaled) into 8 bf16 = one 16B store payload.
__device__ __forceinline__ uint4 pack8(v8f a, float s) {
  union { bf16 h[8]; uint4 u; } r;
#pragma unroll
  for (int i = 0; i < 8; ++i) r.h[i] = (bf16)(a[i] * s);
  return r.u;
}

// ---------------------------------------------------------------------------
// (1) f32 -> bf16 weight conversion
__global__ void __launch_bounds__(256)
convert_kernel(const float* __restrict__ src, bf16* __restrict__ dst, int n) {
  int i = blockIdx.x * 256 + threadIdx.x;
  if (i < n) dst[i] = (bf16)src[i];
}

// ---------------------------------------------------------------------------
// (2) Depthwise 3x3 conv (stride1, pad1) on channel-last x -> qdw bf16 (B,N,C)
__global__ void __launch_bounds__(256)
dwconv_kernel(const float* __restrict__ x, const float* __restrict__ w,
              const float* __restrict__ bias, bf16* __restrict__ qdw) {
  int idx = blockIdx.x * 256 + threadIdx.x;   // over B*N*C = 2^22
  int c = idx & (Cc - 1);
  int n = (idx >> 8) & (Nn - 1);
  int b = idx >> 20;
  int h = n >> 6, wc = n & 63;
  float sum = bias[c];
#pragma unroll
  for (int dh = -1; dh <= 1; ++dh) {
    int hh = h + dh;
    if (hh < 0 || hh >= Hh) continue;
#pragma unroll
    for (int dw = -1; dw <= 1; ++dw) {
      int ww = wc + dw;
      if (ww < 0 || ww >= Ww) continue;
      sum += x[((size_t)b * Nn + hh * Ww + ww) * Cc + c] *
             w[c * 9 + (dh + 1) * 3 + (dw + 1)];
    }
  }
  qdw[idx] = (bf16)sum;
}

// ---------------------------------------------------------------------------
// (3) SR: strided depthwise 3x3 (stride2, pad1, no bias) + LayerNorm over C
//     one block per (b, m); 256 threads = one per channel.
__global__ void __launch_bounds__(256)
srln_kernel(const float* __restrict__ x, const float* __restrict__ w,
            const float* __restrict__ g, const float* __restrict__ beta,
            bf16* __restrict__ xn) {
  __shared__ float red[256];
  int c = threadIdx.x;
  int bm = blockIdx.x;                 // over B*M
  int b = bm >> 10, m = bm & (Mm - 1);
  int mh = m >> 5, mw = m & 31;
  float sum = 0.f;
#pragma unroll
  for (int i = 0; i < 3; ++i) {
    int hh = 2 * mh + i - 1;
    if (hh < 0 || hh >= Hh) continue;
#pragma unroll
    for (int j = 0; j < 3; ++j) {
      int ww = 2 * mw + j - 1;
      if (ww < 0 || ww >= Ww) continue;
      sum += x[((size_t)b * Nn + hh * Ww + ww) * Cc + c] * w[c * 9 + i * 3 + j];
    }
  }
  red[c] = sum;
  __syncthreads();
  for (int off = 128; off > 0; off >>= 1) {
    if (c < off) red[c] += red[c + off];
    __syncthreads();
  }
  float mu = red[0] * (1.0f / Cc);
  __syncthreads();
  float d = sum - mu;
  red[c] = d * d;
  __syncthreads();
  for (int off = 128; off > 0; off >>= 1) {
    if (c < off) red[c] += red[c + off];
    __syncthreads();
  }
  float var = red[0] * (1.0f / Cc);
  float y = d * rsqrtf(var + EPSf) * g[c] + beta[c];
  xn[(size_t)bm * Cc + c] = (bf16)y;
}

// ---------------------------------------------------------------------------
// (4) Pointwise 1x1 GEMM (transposed orientation: A = weight rows) -> Q
//     q[b,h,n,d] bf16, attention scale folded in.
__global__ void __launch_bounds__(128)
pw_gemm_kernel(const bf16* __restrict__ qdw, const bf16* __restrict__ wb,
               const float* __restrict__ bias, bf16* __restrict__ qout) {
  const int wid = threadIdx.x >> 5, lane = threadIdx.x & 31;
  const int t = blockIdx.x * 4 + wid;
  const int ct = (t & 15) * 16;        // c_out tile
  const int st = (t >> 4) * 16;        // spatial tile over B*N
  v8f acc = {};
  for (int kk = 0; kk < Cc; kk += 32) {
    v16bf a = load_tile(wb, Cc, ct, kk);      // weights (c_out x c_in)
    v16bf bop = load_tile(qdw, Cc, st, kk);   // activations as B columns
    acc = wmma_bf16(a, bop, acc);
  }
  int hi = (lane & 16) ? 8 : 0;
#pragma unroll
  for (int j = 0; j < 8; ++j) acc[j] = (acc[j] + bias[ct + hi + j]) * SCALE;
  int s = st + (lane & 15);
  int b = s >> 12, n = s & (Nn - 1);
  int crow = ct + hi;
  int h = crow >> 5, d0 = crow & 31;
  bf16* op = qout + (((size_t)(b * HEADS + h) * Nn + n) * Dd + d0);
  *reinterpret_cast<uint4*>(op) = pack8(acc, 1.0f);
}

// ---------------------------------------------------------------------------
// (5) KV GEMM: xn (B*M, C) x kv_w^T -> K (B,h,M,D) and V stored transposed
//     as vT (B,h,D,M) so it loads as a contiguous A-operand in attention.
__global__ void __launch_bounds__(128)
kv_gemm_kernel(const bf16* __restrict__ xn, const bf16* __restrict__ wb,
               const float* __restrict__ bias, bf16* __restrict__ kout,
               bf16* __restrict__ vT) {
  const int wid = threadIdx.x >> 5, lane = threadIdx.x & 31;
  const int t = blockIdx.x * 4 + wid;
  const int ct = (t & 31) * 16;        // c_out tile over 2C
  const int st = (t >> 5) * 16;        // spatial tile over B*M
  v8f acc = {};
  if (ct < Cc) {  // --- K half: transposed orientation (rows = c_out) ---
    for (int kk = 0; kk < Cc; kk += 32) {
      v16bf a = load_tile(wb, Cc, ct, kk);
      v16bf bop = load_tile(xn, Cc, st, kk);
      acc = wmma_bf16(a, bop, acc);
    }
    int hi = (lane & 16) ? 8 : 0;
#pragma unroll
    for (int j = 0; j < 8; ++j) acc[j] += bias[ct + hi + j];
    int s = st + (lane & 15);
    int b = s >> 10, m = s & (Mm - 1);
    int crow = ct + hi, h = crow >> 5, d0 = crow & 31;
    bf16* op = kout + (((size_t)(b * HEADS + h) * Mm + m) * Dd + d0);
    *reinterpret_cast<uint4*>(op) = pack8(acc, 1.0f);
  } else {        // --- V half: normal orientation (rows = spatial) ---
    for (int kk = 0; kk < Cc; kk += 32) {
      v16bf a = load_tile(xn, Cc, st, kk);
      v16bf bop = load_tile(wb, Cc, ct, kk);
      acc = wmma_bf16(a, bop, acc);
    }
    int c = ct + (lane & 15);
    float bc = bias[c];
#pragma unroll
    for (int j = 0; j < 8; ++j) acc[j] += bc;
    int cv = c - Cc, h = cv >> 5, d = cv & 31;
    int hi = (lane & 16) ? 8 : 0;
    int srow = st + hi;
    int b = srow >> 10, m0 = srow & (Mm - 1);
    bf16* op = vT + (((size_t)(b * HEADS + h) * Dd + d) * Mm + m0);
    *reinterpret_cast<uint4*>(op) = pack8(acc, 1.0f);
  }
}

// ---------------------------------------------------------------------------
// (6) Flash attention. One wave = 16 queries of one (b,h). Computes S^T tiles
//     (K as A, Q^T as B) so the softmaxed P^T feeds the PV WMMA as B operand
//     with zero cross-lane data movement. Online softmax per lane + one
//     shfl_xor(16) per chunk.
__global__ void __launch_bounds__(128)
attn_kernel(const bf16* __restrict__ q, const bf16* __restrict__ k,
            const bf16* __restrict__ vT, bf16* __restrict__ att) {
  const int wid = threadIdx.x >> 5, lane = threadIdx.x & 31;
  const int t = blockIdx.x;
  const int bh = t >> 6;                       // 0..31
  const int n0 = (t & 63) * 64 + wid * 16;     // query tile start
  const int b = bh >> 3, h = bh & 7;

  const bf16* qbase = q + (size_t)bh * Nn * Dd;
  const bf16* kbase = k + (size_t)bh * Mm * Dd;
  const bf16* vbase = vT + (size_t)bh * Dd * Mm;

  // Q tile as B operand: full K=32 (= head dim), loaded once.
  v16bf qb = load_tile(qbase, Dd, n0, 0);

  v8f acc0 = {}, acc1 = {};                    // O^T rows d=0..15 / 16..31
  float m_run = -3.0e38f, l_run = 0.f;
  const v8f zero = {};

  for (int mc = 0; mc < Mm; mc += 32) {
    if (mc + 32 < Mm) {
      __builtin_prefetch(kbase + (size_t)(mc + 32) * Dd, 0, 0);
      __builtin_prefetch(vbase + (mc + 32), 0, 0);
    }
    v16bf ka0 = load_tile(kbase, Dd, mc, 0);       // keys mc..mc+15
    v16bf ka1 = load_tile(kbase, Dd, mc + 16, 0);  // keys mc+16..mc+31
    v8f s0 = wmma_bf16(ka0, qb, zero);             // S^T tile (m x n)
    v8f s1 = wmma_bf16(ka1, qb, zero);

    // row(=key)-dim softmax for this lane's column n
    float cm = s0[0];
#pragma unroll
    for (int j = 1; j < 8; ++j) cm = fmaxf(cm, s0[j]);
#pragma unroll
    for (int j = 0; j < 8; ++j) cm = fmaxf(cm, s1[j]);
    cm = fmaxf(cm, __shfl_xor(cm, 16, 32));
    float m_new = fmaxf(m_run, cm);
    float alpha = __expf(m_run - m_new);
    float p0[8], p1[8], lsum = 0.f;
#pragma unroll
    for (int j = 0; j < 8; ++j) { p0[j] = __expf(s0[j] - m_new); lsum += p0[j]; }
#pragma unroll
    for (int j = 0; j < 8; ++j) { p1[j] = __expf(s1[j] - m_new); lsum += p1[j]; }
    l_run = l_run * alpha + lsum;
    m_run = m_new;
#pragma unroll
    for (int j = 0; j < 8; ++j) { acc0[j] *= alpha; acc1[j] *= alpha; }

    // P^T packs directly into the B-operand layout (no lane shuffles!)
    union { v16bf v; bf16 hh[16]; } pb;
#pragma unroll
    for (int j = 0; j < 8; ++j) {
      pb.hh[j]     = (bf16)p0[j];
      pb.hh[8 + j] = (bf16)p1[j];
    }
    v16bf va0 = load_tile(vbase, Mm, 0, mc);   // V^T rows d=0..15
    v16bf va1 = load_tile(vbase, Mm, 16, mc);  // V^T rows d=16..31
    acc0 = wmma_bf16(va0, pb.v, acc0);
    acc1 = wmma_bf16(va1, pb.v, acc1);
  }

  float l_tot = l_run + __shfl_xor(l_run, 16, 32);
  float inv = 1.0f / l_tot;
  int n = n0 + (lane & 15);
  int dbase = (lane & 16) ? 8 : 0;
  bf16* op = att + ((size_t)(b * Nn + n)) * Cc + h * Dd + dbase;
  *reinterpret_cast<uint4*>(op)      = pack8(acc0, inv);  // d  0..7 / 8..15
  *reinterpret_cast<uint4*>(op + 16) = pack8(acc1, inv);  // d 16..23 / 24..31
}

// ---------------------------------------------------------------------------
// (7) Projection GEMM + bias -> f32 output (B,N,C)
__global__ void __launch_bounds__(128)
proj_gemm_kernel(const bf16* __restrict__ att, const bf16* __restrict__ wb,
                 const float* __restrict__ bias, float* __restrict__ out) {
  const int wid = threadIdx.x >> 5, lane = threadIdx.x & 31;
  const int t = blockIdx.x * 4 + wid;
  const int ct = (t & 15) * 16;
  const int st = (t >> 4) * 16;
  v8f acc = {};
  for (int kk = 0; kk < Cc; kk += 32) {
    v16bf a = load_tile(wb, Cc, ct, kk);
    v16bf bop = load_tile(att, Cc, st, kk);
    acc = wmma_bf16(a, bop, acc);
  }
  int hi = (lane & 16) ? 8 : 0;
  int s = st + (lane & 15);
  float* op = out + (size_t)s * Cc + ct + hi;
  float4 r0, r1;
  r0.x = acc[0] + bias[ct + hi + 0];
  r0.y = acc[1] + bias[ct + hi + 1];
  r0.z = acc[2] + bias[ct + hi + 2];
  r0.w = acc[3] + bias[ct + hi + 3];
  r1.x = acc[4] + bias[ct + hi + 4];
  r1.y = acc[5] + bias[ct + hi + 5];
  r1.z = acc[6] + bias[ct + hi + 6];
  r1.w = acc[7] + bias[ct + hi + 7];
  *reinterpret_cast<float4*>(op)     = r0;
  *reinterpret_cast<float4*>(op + 4) = r1;
}

// ---------------------------------------------------------------------------
extern "C" void kernel_launch(void* const* d_in, const int* in_sizes, int n_in,
                              void* d_out, int out_size, void* d_ws,
                              size_t ws_size, hipStream_t stream) {
  (void)in_sizes; (void)n_in; (void)out_size; (void)ws_size;
  const float* x      = (const float*)d_in[0];
  const float* dw_w   = (const float*)d_in[1];
  const float* dw_b   = (const float*)d_in[2];
  const float* pw_w   = (const float*)d_in[3];
  const float* pw_b   = (const float*)d_in[4];
  const float* sr_w   = (const float*)d_in[5];
  const float* ln_g   = (const float*)d_in[6];
  const float* ln_b   = (const float*)d_in[7];
  const float* kv_w   = (const float*)d_in[8];
  const float* kv_b   = (const float*)d_in[9];
  const float* proj_w = (const float*)d_in[10];
  const float* proj_b = (const float*)d_in[11];
  float* out = (float*)d_out;

  char* ws = (char*)d_ws;
  size_t off = 0;
  auto alloc = [&](size_t bytes) -> char* {
    char* p = ws + off;
    off = (off + bytes + 255) & ~(size_t)255;
    return p;
  };
  bf16* qdw  = (bf16*)alloc((size_t)Bn * Nn * Cc * 2);        // 8 MB
  bf16* q    = (bf16*)alloc((size_t)Bn * Nn * Cc * 2);        // 8 MB
  bf16* xn   = (bf16*)alloc((size_t)Bn * Mm * Cc * 2);        // 2 MB
  bf16* kmat = (bf16*)alloc((size_t)Bn * HEADS * Mm * Dd * 2);// 2 MB
  bf16* vT   = (bf16*)alloc((size_t)Bn * HEADS * Dd * Mm * 2);// 2 MB
  bf16* att  = (bf16*)alloc((size_t)Bn * Nn * Cc * 2);        // 8 MB
  bf16* pwb  = (bf16*)alloc((size_t)Cc * Cc * 2);
  bf16* kvb  = (bf16*)alloc((size_t)2 * Cc * Cc * 2);
  bf16* prb  = (bf16*)alloc((size_t)Cc * Cc * 2);

  convert_kernel<<<(Cc * Cc + 255) / 256, 256, 0, stream>>>(pw_w, pwb, Cc * Cc);
  convert_kernel<<<(2 * Cc * Cc + 255) / 256, 256, 0, stream>>>(kv_w, kvb, 2 * Cc * Cc);
  convert_kernel<<<(Cc * Cc + 255) / 256, 256, 0, stream>>>(proj_w, prb, Cc * Cc);

  dwconv_kernel<<<(Bn * Nn * Cc) / 256, 256, 0, stream>>>(x, dw_w, dw_b, qdw);
  srln_kernel<<<Bn * Mm, 256, 0, stream>>>(x, sr_w, ln_g, ln_b, xn);
  pw_gemm_kernel<<<(16 * (Bn * Nn / 16)) / 4, 128, 0, stream>>>(qdw, pwb, pw_b, q);
  kv_gemm_kernel<<<(32 * (Bn * Mm / 16)) / 4, 128, 0, stream>>>(xn, kvb, kv_b, kmat, vT);
  attn_kernel<<<Bn * HEADS * (Nn / 64), 128, 0, stream>>>(q, kmat, vT, att);
  proj_gemm_kernel<<<(16 * (Bn * Nn / 16)) / 4, 128, 0, stream>>>(att, prb, proj_b, out);
}
